// GCN_44693429682809
// MI455X (gfx1250) — matrix operand order repaired and checked by reading, not compiled, not run on hardware
//
#include <hip/hip_runtime.h>
#include <cstdint>
#include <cstddef>

// ---------------------------------------------------------------------------
// GCN (3x GCNConv) for MI455X / gfx1250.
//
// Roofline: GEMMs are ~1.5 GFLOP total (noise). The aggregation is
// ~1.1 GB of gathers + 1.1 GB of f32 atomic adds, but the per-layer feature
// matrices (12.8 MB) sit in the 192 MB L2, so scatter/gather is L2-resident.
// Strategy: WMMA (split-f16, f32 accumulate) for x@W, per-edge wave-parallel
// global_atomic_add_f32 for aggregation, degree/rsqrt computed once.
// ---------------------------------------------------------------------------

typedef __attribute__((ext_vector_type(16))) _Float16 v16h;
typedef __attribute__((ext_vector_type(8)))  float    v8f;

// ---------------- degree / normalization ----------------
__global__ void zero_f32(float* __restrict__ p, int n) {
  int i = blockIdx.x * blockDim.x + threadIdx.x;
  if (i < n) p[i] = 0.0f;
}

__global__ void degree_accum(const int* __restrict__ dst, const float* __restrict__ ew,
                             float* __restrict__ deg, int E) {
  int i = blockIdx.x * blockDim.x + threadIdx.x;
  if (i < E) atomicAdd(deg + dst[i], ew[i]);
}

__global__ void finalize_dinv(float* __restrict__ dinv, int n) {
  int i = blockIdx.x * blockDim.x + threadIdx.x;
  if (i < n) dinv[i] = rsqrtf(dinv[i] + 1.0f);   // deg >= 1 always (self loop)
}

// ---------------- WMMA GEMM: out[N x FOUT_PAD] = A[N x K] * W[K x FOUT] ----
// One wave per 16x16 output tile. Split-f16 (hi/lo) for ~f32 accuracy:
//   D ~= Ah*Bh + Al*Bh + Ah*Bl   (f32 accumulate)
// A fragment per ISA 7.12.2 (16-bit A 16x32): lane<16 holds K {0..7,16..23},
// lane>=16 holds K {8..15,24..31} of row M=lane&15. B: lane = col N=lane&15, same K sets as A.
// D: VGPR r -> row r + 8*(lane>>4), col lane&15.
template <int K, int FOUT_PAD, int FOUT>
__global__ __launch_bounds__(128) void gemm_wmma(const float* __restrict__ A,
                                                 const float* __restrict__ W,
                                                 float* __restrict__ out, int N) {
  const int colTiles   = FOUT_PAD / 16;
  const int totalTiles = (N / 16) * colTiles;      // N is a multiple of 16
  int tile = blockIdx.x * 4 + (threadIdx.x >> 5);
  if (tile >= totalTiles) return;                  // wave-uniform: EXEC stays all-1
  int lane    = threadIdx.x & 31;
  int rowTile = tile / colTiles;
  int colTile = tile - rowTile * colTiles;
  int half    = lane >> 4;
  int m       = lane & 15;
  int row     = rowTile * 16 + m;
  int n0      = colTile * 16;

  v8f acc = {};
  for (int kt = 0; kt < K; kt += 32) {
    // ---- A fragment (two 32B runs per lane, f32 -> split f16) ----
    const float* ap = A + (size_t)row * K + kt + half * 8;
    float4 q0 = *(const float4*)(ap + 0);
    float4 q1 = *(const float4*)(ap + 4);
    float4 q2 = *(const float4*)(ap + 16);
    float4 q3 = *(const float4*)(ap + 20);
    float av[16] = {q0.x, q0.y, q0.z, q0.w, q1.x, q1.y, q1.z, q1.w,
                    q2.x, q2.y, q2.z, q2.w, q3.x, q3.y, q3.z, q3.w};
    v16h ah, al;
#pragma unroll
    for (int i = 0; i < 16; ++i) {
      _Float16 h = (_Float16)av[i];
      ah[i] = h;
      al[i] = (_Float16)(av[i] - (float)h);
    }
    // ---- B fragment: W col (n0+m), K rows same order as A (zero-pad past FOUT) ----
    const float* wp = W + (size_t)(kt + half * 8) * FOUT + n0 + m;
    v16h bh, bl;
#pragma unroll
    for (int j = 0; j < 16; ++j) {
      int kk = (j < 8) ? j : j + 8;
      float w;
      if (FOUT == FOUT_PAD) w = wp[(size_t)kk * FOUT];
      else                  w = (n0 + m < FOUT) ? wp[(size_t)kk * FOUT] : 0.0f;
      _Float16 h = (_Float16)w;
      bh[j] = h;
      bl[j] = (_Float16)(w - (float)h);
    }
    // (neg_a, A, neg_b, B, c_mod, C, reuse_a, reuse_b)
    acc = __builtin_amdgcn_wmma_f32_16x16x32_f16(false, ah, false, bh, (short)0, acc, false, false);
    acc = __builtin_amdgcn_wmma_f32_16x16x32_f16(false, al, false, bh, (short)0, acc, false, false);
    acc = __builtin_amdgcn_wmma_f32_16x16x32_f16(false, ah, false, bl, (short)0, acc, false, false);
  }
#pragma unroll
  for (int r = 0; r < 8; ++r) {
    int orow = rowTile * 16 + r + 8 * half;
    out[(size_t)orow * FOUT_PAD + n0 + m] = acc[r];
  }
}

// ---------------- self-loop + bias init: out = dinv^2 * xW + b -------------
// out is compact [n x F]; xW has row stride SIN (>= F).
template <int F, int SIN>
__global__ void selfloop_bias(const float* __restrict__ xw, const float* __restrict__ dinv,
                              const float* __restrict__ bias, float* __restrict__ out, int n) {
  int i = blockIdx.x * blockDim.x + threadIdx.x;
  if (i >= n * F) return;
  int r = i / F;
  int c = i - r * F;
  float di = dinv[r];
  out[i] = di * di * xw[(size_t)r * SIN + c] + bias[c];
}

// ---------------- edge aggregation (L2-resident atomics) -------------------
// One wave per edge; lane f covers features f, f+32. Gathers are coalesced
// 128B segments; atomics land on consecutive addresses (L2-resident buffer).
template <int F, int SIN, int SOUT>
__global__ void aggregate(const float* __restrict__ xW, float* __restrict__ agg,
                          const int* __restrict__ src, const int* __restrict__ dst,
                          const float* __restrict__ ew, const float* __restrict__ dinv, int E) {
  int wid = (blockIdx.x * blockDim.x + threadIdx.x) >> 5;
  if (wid >= E) return;
  int lane = threadIdx.x & 31;
  int s = src[wid];
  int d = dst[wid];
  float norm = dinv[s] * ew[wid] * dinv[d];
  const float* xs = xW + (size_t)s * SIN;
  float* ad = agg + (size_t)d * SOUT;
#pragma unroll
  for (int f = lane; f < F; f += 32) atomicAdd(ad + f, norm * xs[f]);
}

__global__ void relu_k(float* __restrict__ p, int n) {
  int i = blockIdx.x * blockDim.x + threadIdx.x;
  if (i < n) p[i] = fmaxf(p[i], 0.0f);
}

// ---------------------------------------------------------------------------
extern "C" void kernel_launch(void* const* d_in, const int* in_sizes, int n_in,
                              void* d_out, int out_size, void* d_ws, size_t ws_size,
                              hipStream_t stream) {
  (void)n_in; (void)out_size; (void)ws_size;
  const float* x  = (const float*)d_in[0];
  const int*   ei = (const int*)d_in[1];
  const float* ew = (const float*)d_in[2];
  const float* W1 = (const float*)d_in[3];
  const float* b1 = (const float*)d_in[4];
  const float* W2 = (const float*)d_in[5];
  const float* b2 = (const float*)d_in[6];
  const float* W3 = (const float*)d_in[7];
  const float* b3 = (const float*)d_in[8];

  const int N = in_sizes[0] / 128;   // 50000 (multiple of 16)
  const int E = in_sizes[2];         // 1.6M
  const int* src = ei;
  const int* dst = ei + E;

  // Workspace: dinv[N] | xw[N*64] (L3 uses stride 48 subset) | hA[N*64] | hB[N*64]
  float* dinv = (float*)d_ws;
  float* xw   = dinv + N;
  float* hA   = xw + (size_t)N * 64;
  float* hB   = hA + (size_t)N * 64;
  float* outp = (float*)d_out;

  const int TB = 256;
  int gN  = (N + TB - 1) / TB;
  int gE  = (E + TB - 1) / TB;
  int gNF = (N * 64 + TB - 1) / TB;
  int gAg = (E + 7) / 8;             // 8 waves/block, 1 edge per wave

  // ---- normalization (shared by all layers): dinv = rsqrt(deg + 1) ----
  zero_f32<<<gN, TB, 0, stream>>>(dinv, N);
  degree_accum<<<gE, TB, 0, stream>>>(dst, ew, dinv, E);
  finalize_dinv<<<gN, TB, 0, stream>>>(dinv, N);

  // ---- layer 1: 128 -> 64, ReLU ----
  {
    int tiles = (N / 16) * (64 / 16);
    gemm_wmma<128, 64, 64><<<(tiles + 3) / 4, 128, 0, stream>>>(x, W1, xw, N);
    selfloop_bias<64, 64><<<gNF, TB, 0, stream>>>(xw, dinv, b1, hA, N);
    aggregate<64, 64, 64><<<gAg, TB, 0, stream>>>(xw, hA, src, dst, ew, dinv, E);
    relu_k<<<gNF, TB, 0, stream>>>(hA, N * 64);
  }
  // ---- layer 2: 64 -> 64, ReLU ----
  {
    int tiles = (N / 16) * (64 / 16);
    gemm_wmma<64, 64, 64><<<(tiles + 3) / 4, 128, 0, stream>>>(hA, W2, xw, N);
    selfloop_bias<64, 64><<<gNF, TB, 0, stream>>>(xw, dinv, b2, hB, N);
    aggregate<64, 64, 64><<<gAg, TB, 0, stream>>>(xw, hB, src, dst, ew, dinv, E);
    relu_k<<<gNF, TB, 0, stream>>>(hB, N * 64);
  }
  // ---- layer 3: 64 -> 40 (padded to 48 in scratch, compact 40 in d_out) ----
  {
    int tiles = (N / 16) * (48 / 16);
    gemm_wmma<64, 48, 40><<<(tiles + 3) / 4, 128, 0, stream>>>(hB, W3, xw, N);
    int g40 = (N * 40 + TB - 1) / TB;
    selfloop_bias<40, 48><<<g40, TB, 0, stream>>>(xw, dinv, b3, outp, N);
    aggregate<40, 48, 40><<<gAg, TB, 0, stream>>>(xw, outp, src, dst, ew, dinv, E);
  }
}
